// BailingMoeV2_5LinearAttention_62620623175750
// MI455X (gfx1250) — matrix operand order repaired and checked by reading, not compiled, not run on hardware
//
#include <hip/hip_runtime.h>
#include <hip/hip_bf16.h>

// ---------------------------------------------------------------------------
// Types for CDNA5 WMMA
// ---------------------------------------------------------------------------
typedef __bf16 bf16_t;
typedef __attribute__((ext_vector_type(16))) __bf16 v16bf;
typedef __attribute__((ext_vector_type(8)))  __bf16 v8bf;
typedef __attribute__((ext_vector_type(8)))  float  v8f;

#define T_TOK 8192
#define H_HEADS 16
#define D_HEAD 128
#define HID 2048
#define QKV_N (3 * H_HEADS * D_HEAD)  // 6144
#define CHUNK 128
#define NCHUNK (T_TOK / CHUNK)        // 64

__device__ __forceinline__ v8f vzero8f() {
  v8f z;
#pragma unroll
  for (int i = 0; i < 8; ++i) z[i] = 0.0f;
  return z;
}

__device__ __forceinline__ v16bf make16(v8bf lo, v8bf hi) {
  v16bf r;
#pragma unroll
  for (int i = 0; i < 8; ++i) { r[i] = lo[i]; r[8 + i] = hi[i]; }
  return r;
}

// A-fragment: 16x32 bf16, row-major source with row stride ld (elements).
// Lane l<16: row M=l, K = {0..7, 16..23}; lane>=16: row M=l-16, K = {8..15, 24..31}.
__device__ __forceinline__ v16bf load_a_frag(const bf16_t* base, int ld) {
  int lane = threadIdx.x & 31;
  const bf16_t* p = base + (lane & 15) * ld + ((lane >> 4) << 3);
  v8bf lo = *(const v8bf*)p;
  v8bf hi = *(const v8bf*)(p + 16);
  return make16(lo, hi);
}

// B-fragment: 32x16 bf16 where the source is stored TRANSPOSED, i.e. source
// row n holds column n of B contiguously in K. Lane l<16: col N=l, K = {0..7, 16..23};
// lane>=16: col N=l-16, K = {8..15, 24..31}.
__device__ __forceinline__ v16bf load_b_frag(const bf16_t* baseT, int ld) {
  int lane = threadIdx.x & 31;
  const bf16_t* p = baseT + (lane & 15) * ld + ((lane >> 4) << 3);
  v8bf lo = *(const v8bf*)p;
  v8bf hi = *(const v8bf*)(p + 16);
  return make16(lo, hi);
}

__device__ __forceinline__ v8f wmma_bf16(v16bf a, v16bf b, v8f c) {
  return __builtin_amdgcn_wmma_f32_16x16x32_bf16(
      false, a, false, b, (short)0, c, false, false);
}

// ---------------------------------------------------------------------------
// 1) fp32 -> bf16 convert
// ---------------------------------------------------------------------------
__global__ __launch_bounds__(256) void cvt_bf16_kernel(
    const float* __restrict__ in, bf16_t* __restrict__ out, int n) {
  int i = blockIdx.x * 256 + threadIdx.x;
  if (i < n) out[i] = (bf16_t)in[i];
}

// 2) transpose + convert: in[K][N] fp32 -> out[N][K] bf16
__global__ __launch_bounds__(256) void transpose_cvt_kernel(
    const float* __restrict__ in, bf16_t* __restrict__ out, int K, int N) {
  long long i = (long long)blockIdx.x * 256 + threadIdx.x;
  if (i >= (long long)K * N) return;
  int k = (int)(i / N);
  int n = (int)(i % N);
  out[(long long)n * K + k] = (bf16_t)in[i];
}

// ---------------------------------------------------------------------------
// 3) Generic WMMA GEMM: C[M][N] f32 = A[M][K] bf16 @ B (given as BT[N][K] bf16)
//    Block = 256 threads = 8 waves; wave w -> rows [blockY*128 + 16w), cols
//    [blockX*64, +64).  Software-pipelined with explicit ping/pong fragment
//    buffers (2x unroll) so there are no inter-buffer register copies: even
//    K-steps compute from buffer 0 while buffer 1 loads, and vice-versa.
// ---------------------------------------------------------------------------
__global__ __launch_bounds__(256) void gemm_bf16_kernel(
    const bf16_t* __restrict__ A, const bf16_t* __restrict__ BT,
    float* __restrict__ C, int M, int N, int K) {
  int wave = threadIdx.x >> 5;
  int lane = threadIdx.x & 31;
  int m0 = blockIdx.y * 128 + wave * 16;
  int n0 = blockIdx.x * 64;
  if (m0 >= M || n0 >= N) return;  // wave-uniform: EXEC stays all-ones

  v8f acc[4];
#pragma unroll
  for (int t = 0; t < 4; ++t) acc[t] = vzero8f();

  const bf16_t* arow = A + (size_t)m0 * K;
  const bf16_t* brow[4];
#pragma unroll
  for (int t = 0; t < 4; ++t) brow[t] = BT + (size_t)(n0 + t * 16) * K;

  const int ksteps = K >> 5;

  // ping buffers: K-step 0
  v16bf a0 = load_a_frag(arow, K);
  v16bf b0[4];
#pragma unroll
  for (int t = 0; t < 4; ++t) b0[t] = load_b_frag(brow[t], K);
  // pong buffers (initialized to silence uninitialized-use; overwritten below)
  v16bf a1 = a0;
  v16bf b1[4];
#pragma unroll
  for (int t = 0; t < 4; ++t) b1[t] = b0[t];

  for (int s = 0; s < ksteps; s += 2) {
    // preload odd step into pong while even step computes
    if (s + 1 < ksteps) {
      int kn = (s + 1) << 5;
      a1 = load_a_frag(arow + kn, K);
#pragma unroll
      for (int t = 0; t < 4; ++t) b1[t] = load_b_frag(brow[t] + kn, K);
    }
    __builtin_prefetch(arow + (s << 5) + 256, 0, 1);  // global_prefetch_b8

#pragma unroll
    for (int t = 0; t < 4; ++t) acc[t] = wmma_bf16(a0, b0[t], acc[t]);

    // preload next even step into ping while odd step computes
    if (s + 2 < ksteps) {
      int kn = (s + 2) << 5;
      a0 = load_a_frag(arow + kn, K);
#pragma unroll
      for (int t = 0; t < 4; ++t) b0[t] = load_b_frag(brow[t] + kn, K);
    }
    if (s + 1 < ksteps) {
#pragma unroll
      for (int t = 0; t < 4; ++t) acc[t] = wmma_bf16(a1, b1[t], acc[t]);
    }
  }

  int rsel = (lane >> 4) << 3;
  int col  = lane & 15;
#pragma unroll
  for (int t = 0; t < 4; ++t)
#pragma unroll
    for (int j = 0; j < 8; ++j)
      C[(size_t)(m0 + rsel + j) * N + n0 + t * 16 + col] = acc[t][j];
}

// ---------------------------------------------------------------------------
// 4) QKV postprocess: RMSNorm(q,k) + RoPE(first 64 dims) + q*D^-0.5 -> bf16
//    One wave per (t, h); lane handles d in {lane, lane+32, lane+64, lane+96}
//    so RoPE pairs (d, d+32) live in the same thread.
// ---------------------------------------------------------------------------
__global__ __launch_bounds__(256) void qkv_post_kernel(
    const float* __restrict__ qkv, const int* __restrict__ positions,
    const float* __restrict__ qw, const float* __restrict__ kw,
    bf16_t* __restrict__ qb, bf16_t* __restrict__ kb, bf16_t* __restrict__ vb) {
  int wave = threadIdx.x >> 5;
  int lane = threadIdx.x & 31;
  int gidx = blockIdx.x * 8 + wave;       // t*H + h
  int t = gidx >> 4;
  int h = gidx & 15;
  size_t base = (size_t)t * QKV_N + (size_t)h * D_HEAD;

  float q[4], k[4], v[4];
  float sq = 0.0f, sk = 0.0f;
#pragma unroll
  for (int i = 0; i < 4; ++i) {
    int d = lane + 32 * i;
    q[i] = qkv[base + d];
    k[i] = qkv[base + 2048 + d];
    v[i] = qkv[base + 4096 + d];
    sq += q[i] * q[i];
    sk += k[i] * k[i];
  }
#pragma unroll
  for (int m = 16; m >= 1; m >>= 1) {
    sq += __shfl_xor(sq, m, 32);
    sk += __shfl_xor(sk, m, 32);
  }
  float rq = rsqrtf(sq * (1.0f / 128.0f) + 1e-6f);
  float rk = rsqrtf(sk * (1.0f / 128.0f) + 1e-6f);
#pragma unroll
  for (int i = 0; i < 4; ++i) {
    int d = lane + 32 * i;
    q[i] *= rq * qw[d];
    k[i] *= rk * kw[d];
  }
  // RoPE on dims [0,64): pair (lane, lane+32), freq index = lane.
  float pos = (float)positions[t];
  float inv = __powf(10000.0f, -(float)lane * (1.0f / 32.0f));
  float ang = pos * inv;
  float cs = __cosf(ang), sn = __sinf(ang);
  float q1 = q[0], q2 = q[1];
  q[0] = q1 * cs - q2 * sn;
  q[1] = q2 * cs + q1 * sn;
  float k1 = k[0], k2 = k[1];
  k[0] = k1 * cs - k2 * sn;
  k[1] = k2 * cs + k1 * sn;

  const float scale = 0.08838834764831845f;  // D^-0.5
  size_t ob = (size_t)t * 2048 + (size_t)h * D_HEAD;
#pragma unroll
  for (int i = 0; i < 4; ++i) {
    int d = lane + 32 * i;
    qb[ob + d] = (bf16_t)(q[i] * scale);
    kb[ob + d] = (bf16_t)k[i];
    vb[ob + d] = (bf16_t)v[i];
  }
}

// ---------------------------------------------------------------------------
// 5) GLA chunk recurrence: one block (8 waves) per head, sequential chunks.
//    LDS: S f32 (64K) | SbT bf16 (32K, transposed) | qs | ks | vts | kts | As
//    A-fragments for each phase are hoisted into registers (loaded once,
//    reused across all 8 column tiles) -> 8x fewer ds loads per WMMA.
// ---------------------------------------------------------------------------
__global__ __launch_bounds__(256) void gla_kernel(
    const bf16_t* __restrict__ qb, const bf16_t* __restrict__ kb,
    const bf16_t* __restrict__ vb, const float* __restrict__ rs,
    float* __restrict__ o_out, float* __restrict__ state_out) {
  extern __shared__ char smem[];
  float*  S   = (float*)smem;                        // 128x128 f32
  bf16_t* SbT = (bf16_t*)(smem + 65536);             // S^T bf16
  bf16_t* qs  = SbT + 16384;                         // q chunk, row-major
  bf16_t* ks  = qs + 16384;                          // k chunk, row-major
  bf16_t* vts = ks + 16384;                          // v^T
  bf16_t* kts = vts + 16384;                         // (k*kdec)^T
  bf16_t* As  = kts + 16384;                         // decayed A

  int h = blockIdx.x;
  int tid = threadIdx.x;
  int wave = tid >> 5, lane = tid & 31;
  int rsel = (lane >> 4) << 3;
  int col  = lane & 15;

  // slopes: -(1 - 12/31 + 1e-5) * 2^(-0.5*(h+1))
  float gamma = -(1.0f - 12.0f / 31.0f + 1e-5f) * exp2f(-0.5f * (float)(h + 1));
  float sdec = __expf(gamma * (float)CHUNK);

  for (int idx = tid; idx < 16384; idx += 256) S[idx] = rs[h * 16384 + idx];
  __syncthreads();
  for (int idx = tid; idx < 16384; idx += 256) {
    int d = idx >> 7, e = idx & 127;
    SbT[e * 128 + d] = (bf16_t)S[idx];
  }
  __syncthreads();

  const int i = wave;  // row-tile owned by this wave

  for (int c = 0; c < NCHUNK; ++c) {
    // stage chunk tiles
    for (int idx = tid; idx < 16384; idx += 256) {
      int r = idx >> 7, d = idx & 127;
      size_t g = (size_t)(c * CHUNK + r) * 2048 + (size_t)h * D_HEAD + d;
      bf16_t qv = qb[g], kv = kb[g], vv = vb[g];
      qs[r * 128 + d] = qv;
      ks[r * 128 + d] = kv;
      vts[d * 128 + r] = vv;
      kts[d * 128 + r] = (bf16_t)((float)kv * __expf(gamma * (float)(127 - r)));
    }
    __syncthreads();

    // hoisted q A-fragments for this wave's 16-row strip (used in A and O)
    v16bf qa[4];
#pragma unroll
    for (int kk = 0; kk < 4; ++kk) qa[kk] = load_a_frag(qs + i * 2048 + kk * 32, 128);

    // Phase A: A = (q k^T) * decay-mask
    for (int j = 0; j < 8; ++j) {
      v8f acc = vzero8f();
#pragma unroll
      for (int kk = 0; kk < 4; ++kk) {
        v16bf b = load_b_frag(ks + j * 2048 + kk * 32, 128);
        acc = wmma_bf16(qa[kk], b, acc);
      }
#pragma unroll
      for (int jj = 0; jj < 8; ++jj) {
        int r = i * 16 + rsel + jj;
        int s = j * 16 + col;
        float v = (r >= s) ? acc[jj] * __expf(gamma * (float)(r - s)) : 0.0f;
        As[r * 128 + s] = (bf16_t)v;
      }
    }
    __syncthreads();

    // Phase O: o = A v + diag(qdec) (q S)
    {
      v16bf aa[4];
#pragma unroll
      for (int kk = 0; kk < 4; ++kk) aa[kk] = load_a_frag(As + i * 2048 + kk * 32, 128);
      for (int j = 0; j < 8; ++j) {
        v8f acc = vzero8f();
        v8f acc2 = vzero8f();
#pragma unroll
        for (int kk = 0; kk < 4; ++kk) {
          v16bf bv = load_b_frag(vts + j * 2048 + kk * 32, 128);
          acc = wmma_bf16(aa[kk], bv, acc);
          v16bf bs = load_b_frag(SbT + j * 2048 + kk * 32, 128);
          acc2 = wmma_bf16(qa[kk], bs, acc2);
        }
#pragma unroll
        for (int jj = 0; jj < 8; ++jj) {
          int r = i * 16 + rsel + jj;
          float val = acc[jj] + acc2[jj] * __expf(gamma * (float)(r + 1));
          o_out[(size_t)(c * CHUNK + r) * 2048 + (size_t)h * D_HEAD + j * 16 + col] = val;
        }
      }
    }
    __syncthreads();

    // Phase S: S = sdec*S + (k kdec)^T v
    {
      v16bf ka[4];
#pragma unroll
      for (int kk = 0; kk < 4; ++kk) ka[kk] = load_a_frag(kts + i * 2048 + kk * 32, 128);
      for (int j = 0; j < 8; ++j) {
        v8f acc = vzero8f();
#pragma unroll
        for (int kk = 0; kk < 4; ++kk) {
          v16bf b = load_b_frag(vts + j * 2048 + kk * 32, 128);
          acc = wmma_bf16(ka[kk], b, acc);
        }
#pragma unroll
        for (int jj = 0; jj < 8; ++jj) {
          int d = i * 16 + rsel + jj;
          int e = j * 16 + col;
          float snv = sdec * S[d * 128 + e] + acc[jj];
          S[d * 128 + e] = snv;
          SbT[e * 128 + d] = (bf16_t)snv;
        }
      }
    }
    __syncthreads();
  }

  for (int idx = tid; idx < 16384; idx += 256)
    state_out[h * 16384 + idx] = S[idx];
}

// ---------------------------------------------------------------------------
// 6) Group-RMSNorm * g_norm_w * sigmoid(gate) -> bf16. One wave per group.
// ---------------------------------------------------------------------------
__global__ __launch_bounds__(256) void gn_gate_kernel(
    const float* __restrict__ o, const float* __restrict__ gate_lin,
    const float* __restrict__ gw, bf16_t* __restrict__ ag) {
  int t = blockIdx.x;
  int wave = threadIdx.x >> 5, lane = threadIdx.x & 31;
  size_t base = (size_t)t * 2048 + wave * 256;
  float x[8];
  float ss = 0.0f;
#pragma unroll
  for (int i = 0; i < 8; ++i) {
    x[i] = o[base + lane + 32 * i];
    ss += x[i] * x[i];
  }
#pragma unroll
  for (int m = 16; m >= 1; m >>= 1) ss += __shfl_xor(ss, m, 32);
  float r = rsqrtf(ss * (1.0f / 256.0f) + 1e-6f);
#pragma unroll
  for (int i = 0; i < 8; ++i) {
    int cg = wave * 256 + lane + 32 * i;
    float g = gate_lin[(size_t)t * 2048 + cg];
    float sig = 1.0f / (1.0f + __expf(-g));
    ag[(size_t)t * 2048 + cg] = (bf16_t)(x[i] * r * gw[cg] * sig);
  }
}

// ---------------------------------------------------------------------------
// Host launcher
// ---------------------------------------------------------------------------
extern "C" void kernel_launch(void* const* d_in, const int* in_sizes, int n_in,
                              void* d_out, int out_size, void* d_ws, size_t ws_size,
                              hipStream_t stream) {
  (void)in_sizes; (void)n_in; (void)out_size; (void)ws_size;

  const int*   positions = (const int*)d_in[0];
  const float* hidden    = (const float*)d_in[1];
  const float* rstate    = (const float*)d_in[2];
  const float* w_qkv     = (const float*)d_in[3];
  const float* w_g       = (const float*)d_in[4];
  const float* w_dense   = (const float*)d_in[5];
  const float* q_norm_w  = (const float*)d_in[6];
  const float* k_norm_w  = (const float*)d_in[7];
  const float* g_norm_w  = (const float*)d_in[8];
  float* out = (float*)d_out;

  char* ws = (char*)d_ws;
  size_t off = 0;
  auto take = [&](size_t bytes) {
    size_t r = off;
    off += (bytes + 255) & ~(size_t)255;
    return r;
  };
  bf16_t* hb   = (bf16_t*)(ws + take((size_t)T_TOK * HID * 2));
  bf16_t* wqT  = (bf16_t*)(ws + take((size_t)HID * QKV_N * 2));
  bf16_t* wgT  = (bf16_t*)(ws + take((size_t)HID * HID * 2));
  bf16_t* wdT  = (bf16_t*)(ws + take((size_t)HID * HID * 2));
  float*  qkv  = (float*)(ws + take((size_t)T_TOK * QKV_N * 4));
  float*  glin = (float*)(ws + take((size_t)T_TOK * HID * 4));
  bf16_t* qb   = (bf16_t*)(ws + take((size_t)T_TOK * HID * 2));
  bf16_t* kb   = (bf16_t*)(ws + take((size_t)T_TOK * HID * 2));
  bf16_t* vb   = (bf16_t*)(ws + take((size_t)T_TOK * HID * 2));
  float*  oo   = (float*)(ws + take((size_t)T_TOK * HID * 4));
  bf16_t* ag   = (bf16_t*)(ws + take((size_t)T_TOK * HID * 2));

  // 1) convert activations
  {
    int n = T_TOK * HID;
    cvt_bf16_kernel<<<(n + 255) / 256, 256, 0, stream>>>(hidden, hb, n);
  }
  // 2) transpose-convert weights to [N][K] bf16
  {
    long long n = (long long)HID * QKV_N;
    transpose_cvt_kernel<<<(unsigned)((n + 255) / 256), 256, 0, stream>>>(w_qkv, wqT, HID, QKV_N);
    long long m = (long long)HID * HID;
    transpose_cvt_kernel<<<(unsigned)((m + 255) / 256), 256, 0, stream>>>(w_g, wgT, HID, HID);
    transpose_cvt_kernel<<<(unsigned)((m + 255) / 256), 256, 0, stream>>>(w_dense, wdT, HID, HID);
  }
  // 3) qkv and gate projections
  gemm_bf16_kernel<<<dim3(QKV_N / 64, T_TOK / 128), 256, 0, stream>>>(
      hb, wqT, qkv, T_TOK, QKV_N, HID);
  gemm_bf16_kernel<<<dim3(HID / 64, T_TOK / 128), 256, 0, stream>>>(
      hb, wgT, glin, T_TOK, HID, HID);
  // 4) RMSNorm + RoPE + scale
  qkv_post_kernel<<<(T_TOK * H_HEADS) / 8, 256, 0, stream>>>(
      qkv, positions, q_norm_w, k_norm_w, qb, kb, vb);
  // 5) GLA recurrence (writes state tail of d_out)
  {
    size_t shmem = 65536 + 6 * 32768;  // 256 KB dynamic LDS (<= 320 KB/WGP)
    gla_kernel<<<H_HEADS, 256, shmem, stream>>>(
        qb, kb, vb, rstate, oo, out + (size_t)T_TOK * HID);
  }
  // 6) group norm + gate
  gn_gate_kernel<<<T_TOK, 256, 0, stream>>>(oo, glin, g_norm_w, ag);
  // 7) output projection -> d_out
  gemm_bf16_kernel<<<dim3(HID / 64, T_TOK / 128), 256, 0, stream>>>(
      ag, wdT, out, T_TOK, HID, HID);
}
